// VeryAggressiveContrastiveLoss_40175124087460
// MI455X (gfx1250) — matrix operand order repaired and checked by reading, not compiled, not run on hardware
//
#include <hip/hip_runtime.h>
#include <hip/hip_bf16.h>
#include <stdint.h>

// ---------------------------------------------------------------------------
// Supervised contrastive loss, B=8192, D=256, A=4, T=0.1
//   sim = (E E^T)/T  computed as split-bf16 WMMA (hi*hi + hi*lo + lo*hi)
//   fused exp / diagonal-skip / label-match / row-sum epilogue per 16x16 tile
// ---------------------------------------------------------------------------

typedef __attribute__((ext_vector_type(16))) __bf16 v16bf;
typedef __attribute__((ext_vector_type(8)))  float  v8f;

#define B_N   8192
#define D_K   256
#define T_INV 10.0f

// workspace byte offsets (total ~8.53 MB)
#define OFF_EHI  0u
#define OFF_ELO  (4u * 1024u * 1024u)
#define OFF_PACK (8u * 1024u * 1024u)
#define OFF_POSP (OFF_PACK + 32768u)
#define OFF_ALLP (OFF_POSP + 8u * 8192u * 4u)

static __device__ __forceinline__ unsigned short f2bf(float f) {
  // round-to-nearest-even f32 -> bf16 (inputs are finite)
  unsigned u = __float_as_uint(f);
  return (unsigned short)((u + 0x7fffu + ((u >> 16) & 1u)) >> 16);
}
static __device__ __forceinline__ float bf2f(unsigned short h) {
  return __uint_as_float(((unsigned)h) << 16);
}

// --------------------------- prep: bf16 split + packed labels ---------------
__global__ void prep_kernel(const float* __restrict__ emb,
                            const int* __restrict__ labels,
                            unsigned short* __restrict__ Ehi,
                            unsigned short* __restrict__ Elo,
                            unsigned* __restrict__ packed) {
  int idx = blockIdx.x * 256 + threadIdx.x;
  if (idx < B_N * D_K) {
    float x = emb[idx];
    unsigned short h = f2bf(x);
    unsigned short l = f2bf(x - bf2f(h));
    Ehi[idx] = h;
    Elo[idx] = l;
  }
  if (idx < B_N) {
    // label values in [0,6): pack 4 aspects into one u32 (one byte each)
    const int* lp = labels + idx * 4;
    unsigned p = (unsigned)(lp[0] & 0xff) | ((unsigned)(lp[1] & 0xff) << 8) |
                 ((unsigned)(lp[2] & 0xff) << 16) | ((unsigned)(lp[3] & 0xff) << 24);
    packed[idx] = p;
  }
}

union Frag { uint4 q[2]; v16bf v; };

#define LDS_PITCH 264  // 256 + 8 ushort pad -> row stride 132 words (== 4 mod 64 banks)

// --------------------------- main: fused GEMM + epilogue --------------------
__global__ __launch_bounds__(256) void
simloss_main(const unsigned short* __restrict__ Ehi,
             const unsigned short* __restrict__ Elo,
             const unsigned* __restrict__ packed,
             float* __restrict__ posP, float* __restrict__ allP) {
  __shared__ __align__(16) unsigned short sB[2][2][16 * LDS_PITCH];  // [buf][hi/lo]

  const int tid  = threadIdx.x;
  const int lane = tid & 31;
  const int wave = tid >> 5;
  const int nl   = lane & 15;   // N position within tile (and M row for A frag)
  const int hi   = lane >> 4;   // lane half

  const int rowBase  = blockIdx.x * 128 + wave * 16;  // this wave's 16 rows
  const int colChunk = blockIdx.y * 1024;             // 64 column tiles per chunk

  // ---- A panel (16 rows x K=256, hi+lo) resident in VGPRs, loaded once ----
  // bf16 A-frag layout (16x32): lane half selects 16 contiguous K at offset 16*hi
  const int mA = rowBase + nl;
  Frag ahi[8], alo[8];
#pragma unroll
  for (int k = 0; k < 8; ++k) {
    int o0 = mA * D_K + 32 * k + 16 * hi;
    ahi[k].q[0] = *(const uint4*)(Ehi + o0);
    ahi[k].q[1] = *(const uint4*)(Ehi + o0 + 8);
    alo[k].q[0] = *(const uint4*)(Elo + o0);
    alo[k].q[1] = *(const uint4*)(Elo + o0 + 8);
  }

  // packed labels for this lane's C rows: m = rowBase + 8*hi + v
  unsigned rp[8];
#pragma unroll
  for (int v = 0; v < 8; ++v) rp[v] = packed[rowBase + 8 * hi + v];

  float accAll[8], accPos[8];
#pragma unroll
  for (int v = 0; v < 8; ++v) { accAll[v] = 0.f; accPos[v] = 0.f; }

  // ---- cooperative load of first B panel (16 cols x 256 K, hi+lo) ----
  {
    const int colBase = colChunk;
#pragma unroll
    for (int i = 0; i < 2; ++i) {
      int c = tid + 256 * i;
      int r = c >> 5, k8 = (c & 31) * 8;
      *(uint4*)(&sB[0][0][r * LDS_PITCH + k8]) = *(const uint4*)(Ehi + (colBase + r) * D_K + k8);
      *(uint4*)(&sB[0][1][r * LDS_PITCH + k8]) = *(const uint4*)(Elo + (colBase + r) * D_K + k8);
    }
  }
  __syncthreads();

  for (int ct = 0; ct < 64; ++ct) {
    const int buf = ct & 1;

    // prefetch next panel into the other buffer (global -> reg -> ds_store)
    if (ct + 1 < 64) {
      const int colBase = colChunk + (ct + 1) * 16;
#pragma unroll
      for (int i = 0; i < 2; ++i) {
        int c = tid + 256 * i;
        int r = c >> 5, k8 = (c & 31) * 8;
        *(uint4*)(&sB[buf ^ 1][0][r * LDS_PITCH + k8]) = *(const uint4*)(Ehi + (colBase + r) * D_K + k8);
        *(uint4*)(&sB[buf ^ 1][1][r * LDS_PITCH + k8]) = *(const uint4*)(Elo + (colBase + r) * D_K + k8);
      }
    }

    const int colBase = colChunk + ct * 16;
    const int n  = colBase + nl;
    const unsigned pn = packed[n];

    v8f c = {};
#pragma unroll
    for (int k = 0; k < 8; ++k) {
      // bf16 B-frag (32x16): lane holds 16 contiguous K values of column n
      const unsigned short* ph = &sB[buf][0][nl * LDS_PITCH + 32 * k + 16 * hi];
      const unsigned short* pl = &sB[buf][1][nl * LDS_PITCH + 32 * k + 16 * hi];
      Frag bh, bl;
      bh.q[0] = *(const uint4*)(ph);  bh.q[1] = *(const uint4*)(ph + 8);
      bl.q[0] = *(const uint4*)(pl);  bl.q[1] = *(const uint4*)(pl + 8);
      // split-precision: hi*hi + hi*lo + lo*hi  (lo*lo ~2^-16, dropped)
      c = __builtin_amdgcn_wmma_f32_16x16x32_bf16(false, ahi[k].v, false, bh.v, (short)0, c, false, false);
      c = __builtin_amdgcn_wmma_f32_16x16x32_bf16(false, ahi[k].v, false, bl.v, (short)0, c, false, false);
      c = __builtin_amdgcn_wmma_f32_16x16x32_bf16(false, alo[k].v, false, bh.v, (short)0, c, false, false);
    }

    // fused epilogue: exp, diagonal skip, any-aspect label match, row sums
#pragma unroll
    for (int v = 0; v < 8; ++v) {
      float e = __expf(c[v] * T_INV);
      int m = rowBase + 8 * hi + v;
      unsigned x = rp[v] ^ pn;  // byte-equality trick: any zero byte => match
      bool match = ((x - 0x01010101u) & ~x & 0x80808080u) != 0u;
      if (m != n) {
        accAll[v] += e;
        if (match) accPos[v] += e;
      }
    }
    __syncthreads();
  }

  // reduce over the 16 N-lanes of each half; lane nl==0 of each half writes
#pragma unroll
  for (int v = 0; v < 8; ++v) {
    float a = accAll[v], p = accPos[v];
    for (int off = 1; off < 16; off <<= 1) {
      a += __shfl_xor(a, off, 32);
      p += __shfl_xor(p, off, 32);
    }
    if (nl == 0) {
      int row = rowBase + 8 * hi + v;
      posP[blockIdx.y * B_N + row] = p;  // deterministic: unique (chunk,row) owner
      allP[blockIdx.y * B_N + row] = a;
    }
  }
}

// --------------------------- final scalar reduction -------------------------
__global__ void loss_kernel(const float* __restrict__ posP,
                            const float* __restrict__ allP,
                            float* __restrict__ out) {
  __shared__ float sS[256], sC[256];
  int tid = threadIdx.x;
  float s = 0.f, cnt = 0.f;
  for (int row = tid; row < B_N; row += 256) {
    float p = 0.f, a = 0.f;
#pragma unroll
    for (int j = 0; j < 8; ++j) { p += posP[j * B_N + row]; a += allP[j * B_N + row]; }
    if (p > 0.f) { s -= __logf(p / (a + 1e-8f)); cnt += 1.f; }
  }
  sS[tid] = s; sC[tid] = cnt;
  __syncthreads();
  for (int off = 128; off > 0; off >>= 1) {
    if (tid < off) { sS[tid] += sS[tid + off]; sC[tid] += sC[tid + off]; }
    __syncthreads();
  }
  if (tid == 0) {
    float np = sC[0];
    out[0] = (np > 0.f) ? (sS[0] / fmaxf(np, 1.f)) : 0.f;
  }
}

// --------------------------- launcher ---------------------------------------
extern "C" void kernel_launch(void* const* d_in, const int* in_sizes, int n_in,
                              void* d_out, int out_size, void* d_ws, size_t ws_size,
                              hipStream_t stream) {
  const float* emb    = (const float*)d_in[0];   // [8192,256] f32
  const int*   labels = (const int*)d_in[1];     // [8192,4] int (per harness int convention)
  float* out = (float*)d_out;
  char*  ws  = (char*)d_ws;

  unsigned short* Ehi    = (unsigned short*)(ws + OFF_EHI);
  unsigned short* Elo    = (unsigned short*)(ws + OFF_ELO);
  unsigned*       packed = (unsigned*)(ws + OFF_PACK);
  float*          posP   = (float*)(ws + OFF_POSP);
  float*          allP   = (float*)(ws + OFF_ALLP);

  prep_kernel<<<(B_N * D_K) / 256, 256, 0, stream>>>(emb, labels, Ehi, Elo, packed);

  dim3 grid(B_N / 128, 8);  // 64 row-strips x 8 column-chunks
  simloss_main<<<grid, 256, 0, stream>>>(Ehi, Elo, packed, posP, allP);

  loss_kernel<<<1, 256, 0, stream>>>(posP, allP, out);
}
